// Dense_conv_18107582120619
// MI455X (gfx1250) — matrix operand order repaired and checked by reading, not compiled, not run on hardware
//
#include <hip/hip_runtime.h>

// MI455X (gfx1250, wave32) fused DGCNN dense-conv block.
// Pipeline (all on `stream`, stream-ordered so the 16MB S buffer is reused per batch):
//   1) prep_x:    x (B,C,N) f32 -> xh[b][n][c] f16 (point-major) + xx[b][n] = ||x||^2
//   2) prep_w:    W1/W2/W3 f32 -> f16
//   3) per batch: dist_gemm (WMMA f16->f32, S = X X^T; each wave does a 16x64 strip
//                 reusing A fragments across 4 column tiles -> 8 WMMAs/wave)
//                 topk      (one WAVE per row: per-lane local top-16 over a 64-elem
//                 stride, then 16-round shfl_xor argmax merge -> 2048 waves/batch)
//   4) fused_dense_conv: per point, gather 16 neighbors, conv1+ReLU, conv2+ReLU,
//                 conv3, k-max, all via v_wmma_f32_16x16x32_f16 (M=out ch, N=16 nbrs,
//                 K=32-ch chunks), LDS staging for layer-to-layer transpose.

typedef __attribute__((ext_vector_type(16))) _Float16 v16h;
typedef __attribute__((ext_vector_type(8)))  _Float16 v8h;
typedef __attribute__((ext_vector_type(8)))  float    v8f;

#define DEV static __device__ __forceinline__

constexpr int Bn = 8, Cc = 64, Nn = 2048, Kk = 16, Gg = 64;
constexpr int CHO = Cc + 3 * Gg;   // 256 output channels

// ---------------- workspace layout (bytes) ----------------
constexpr size_t OFF_XH  = 0;
constexpr size_t SZ_XH   = (size_t)Bn * Nn * Cc * 2;          // 2 MB
constexpr size_t OFF_XX  = OFF_XH + SZ_XH;
constexpr size_t SZ_XX   = (size_t)Bn * Nn * 4;               // 64 KB
constexpr size_t OFF_W1  = OFF_XX + SZ_XX;
constexpr size_t SZ_W1   = (size_t)Gg * (2 * Cc) * 2;         // 16 KB
constexpr size_t OFF_W2  = OFF_W1 + SZ_W1;
constexpr size_t SZ_W2   = (size_t)Gg * (Cc + Gg) * 2;        // 16 KB
constexpr size_t OFF_W3  = OFF_W2 + SZ_W2;
constexpr size_t SZ_W3   = (size_t)Gg * (Cc + 2 * Gg) * 2;    // 24 KB
constexpr size_t OFF_IDX = ((OFF_W3 + SZ_W3 + 255) / 256) * 256;
constexpr size_t SZ_IDX  = (size_t)Bn * Nn * Kk * 4;          // 1 MB
constexpr size_t OFF_S   = ((OFF_IDX + SZ_IDX + 255) / 256) * 256;
// S: Nn*Nn*4 = 16 MB (reused per batch, L2-resident on 192MB L2)

// ---------------- fragment helpers ----------------
DEV v16h pack16(v8h lo, v8h hi) {
  v16h t;
#pragma unroll
  for (int i = 0; i < 8; ++i) { t[i] = lo[i]; t[8 + i] = hi[i]; }
  return t;
}

// A fragment 16x32 f16: row = lane%16; lanes0-15: K 0..15, lanes16-31: K 16..31.
DEV v16h load_a16(const _Float16* base, int ld, int k0, int lane) {
  const _Float16* p = base + (size_t)(lane & 15) * ld + k0 + ((lane & 16) ? 16 : 0);
  return pack16(*(const v8h*)p, *(const v8h*)(p + 8));
}

// B fragment 32x16 f16: column = lane%16 (caller gives that column's channel base);
// lanes0-15 hold K=k0..k0+15, lanes16-31 hold K=k0+16..k0+31 (contiguous 32B).
DEV v16h load_b16(const _Float16* colbase, int k0, int lane) {
  const _Float16* p = colbase + k0 + ((lane & 16) ? 16 : 0);
  return pack16(*(const v8h*)p, *(const v8h*)(p + 8));
}

// B fragment of edge-feature difference: nb[c] - ctr[c]
DEV v16h load_b16_diff(const _Float16* nb, const _Float16* ct, int k0, int lane) {
  int o = k0 + ((lane & 16) ? 16 : 0);
  v8h lo = *(const v8h*)(nb + o)     - *(const v8h*)(ct + o);
  v8h hi = *(const v8h*)(nb + o + 8) - *(const v8h*)(ct + o + 8);
  return pack16(lo, hi);
}

DEV v8f wmma_f16(v16h a, v16h b, v8f c) {
  return __builtin_amdgcn_wmma_f32_16x16x32_f16(false, a, false, b, (short)0, c,
                                                false, false);
}

// Max over N (=k neighbors) of a C fragment: N = lane%16, so reduce within each
// 16-lane half with shfl_xor 1/2/4/8; lanes 0 and 16 write channels m and m+8.
DEV void warp_max_store(v8f c, float* outb, int chbase, int lane) {
#pragma unroll
  for (int r = 0; r < 8; ++r) {
    float v = c[r];
    v = fmaxf(v, __shfl_xor(v, 1, 32));
    v = fmaxf(v, __shfl_xor(v, 2, 32));
    v = fmaxf(v, __shfl_xor(v, 4, 32));
    v = fmaxf(v, __shfl_xor(v, 8, 32));
    if ((lane & 15) == 0) {
      int ch = chbase + r + ((lane & 16) ? 8 : 0);
      outb[(size_t)ch * Nn] = v;
    }
  }
}

// ---------------- prep kernels ----------------
__global__ void __launch_bounds__(256)
prep_x_kernel(const float* __restrict__ x, _Float16* __restrict__ xh,
              float* __restrict__ xx) {
  int t = blockIdx.x * 256 + threadIdx.x;            // 0..B*N-1
  int b = t >> 11, n = t & (Nn - 1);
  const float* xp = x + ((size_t)b * Cc) * Nn + n;
  _Float16* o = xh + (size_t)t * Cc;
  float s = 0.0f;
#pragma unroll 8
  for (int c = 0; c < Cc; ++c) {
    float v = xp[(size_t)c * Nn];
    s += v * v;
    o[c] = (_Float16)v;
  }
  xx[t] = s;
}

__global__ void __launch_bounds__(256)
prep_w_kernel(const float* __restrict__ W1, const float* __restrict__ W2,
              const float* __restrict__ W3, _Float16* __restrict__ w1h,
              _Float16* __restrict__ w2h, _Float16* __restrict__ w3h) {
  int t = blockIdx.x * 256 + threadIdx.x;
  if (t < 8192)        w1h[t]         = (_Float16)W1[t];
  else if (t < 16384)  w2h[t - 8192]  = (_Float16)W2[t - 8192];
  else if (t < 28672)  w3h[t - 16384] = (_Float16)W3[t - 16384];
}

// ---------------- distance GEMM: S = X X^T (per batch) ----------------
// Each wave computes a 16x64 strip: A fragments loaded once, reused across
// 4 column tiles -> 8 back-to-back WMMAs per wave, 4x less A-fragment traffic.
__global__ void __launch_bounds__(256)
dist_gemm_kernel(const _Float16* __restrict__ xh, float* __restrict__ S, int b) {
  const int lane = threadIdx.x & 31;
  const int wave = threadIdx.x >> 5;
  const int strip = blockIdx.x * 8 + wave;   // 128 row-tiles x 32 col-groups
  const int rt = strip >> 5, cg = strip & 31;
  const _Float16* xb = xh + (size_t)b * Nn * Cc;
  const _Float16* arow = xb + (size_t)(rt * 16) * Cc;
  const v16h a0 = load_a16(arow, Cc, 0, lane);
  const v16h a1 = load_a16(arow, Cc, 32, lane);
  const int hi8 = (lane & 16) ? 8 : 0;
  const int nn = lane & 15;
#pragma unroll
  for (int j = 0; j < 4; ++j) {
    const int c0 = cg * 64 + j * 16;
    const _Float16* cb = xb + (size_t)(c0 + nn) * Cc;
    v8f c = {};
    c = wmma_f16(a0, load_b16(cb, 0, lane), c);
    c = wmma_f16(a1, load_b16(cb, 32, lane), c);
#pragma unroll
    for (int r = 0; r < 8; ++r)
      S[(size_t)(rt * 16 + hi8 + r) * Nn + c0 + nn] = c[r];
  }
}

// ---------------- top-16 neighbors: one wave per row ----------------
// Lane scans a 64-element stride keeping a local top-16 (global top-16 is a
// subset of the union of per-lane top-16s), then 16 rounds of wave-wide argmax
// with deterministic lane tie-breaking select the final indices.
__global__ void __launch_bounds__(256)
topk_kernel(const float* __restrict__ S, const float* __restrict__ xx,
            int* __restrict__ idx, int b) {
  const int lane = threadIdx.x & 31;
  const int wave = threadIdx.x >> 5;
  const int n = blockIdx.x * 8 + wave;               // row, 0..2047
  const float* xxb = xx + (size_t)b * Nn;
  const float xxn = xxb[n];
  const float* srow = S + (size_t)n * Nn;

  float lv[Kk];
  int li[Kk];
#pragma unroll
  for (int j = 0; j < Kk; ++j) { lv[j] = -3.0e38f; li[j] = 0; }
  for (int m = lane; m < Nn; m += 32) {
    float pd = 2.0f * srow[m] - xxn - xxb[m];        // negative sq distance
    if (pd > lv[Kk - 1]) {
      int j = Kk - 1;
      while (j > 0 && lv[j - 1] < pd) {
        lv[j] = lv[j - 1];
        li[j] = li[j - 1];
        --j;
      }
      lv[j] = pd;
      li[j] = m;
    }
  }
  // merge: each lane exposes the head of its sorted list; 16 argmax rounds
  int h = 0;
  int* o = idx + ((size_t)b * Nn + n) * Kk;
#pragma unroll 1
  for (int j = 0; j < Kk; ++j) {
    float bv = (h < Kk) ? lv[h] : -3.4e38f;
    int bi = (h < Kk) ? li[h] : 0;
    int bl = lane;
#pragma unroll
    for (int md = 16; md >= 1; md >>= 1) {
      float ov = __shfl_xor(bv, md, 32);
      int oi = __shfl_xor(bi, md, 32);
      int ol = __shfl_xor(bl, md, 32);
      if (ov > bv || (ov == bv && ol < bl)) { bv = ov; bi = oi; bl = ol; }
    }
    if (lane == bl) ++h;                             // winner advances its head
    if (lane == 0) o[j] = bi;
  }
}

// ---------------- fused conv stack + k-max ----------------
__global__ void __launch_bounds__(256)
fused_dense_conv_kernel(const _Float16* __restrict__ xh, const int* __restrict__ idx,
                        const _Float16* __restrict__ w1, const _Float16* __restrict__ w2,
                        const _Float16* __restrict__ w3,
                        const float* __restrict__ b1, const float* __restrict__ b2,
                        const float* __restrict__ b3,
                        const float* __restrict__ x, float* __restrict__ out) {
  __shared__ __align__(16) _Float16 y1s[8][Kk][Cc];  // relu1, per-wave tile [k][ch]
  __shared__ __align__(16) _Float16 y2s[8][Kk][Cc];  // relu2
  const int lane = threadIdx.x & 31;
  const int wave = threadIdx.x >> 5;
  const int pt = blockIdx.x * 8 + wave;              // 0..B*N-1 (exact, no tail)
  const int b = pt >> 11;
  const int n = pt & (Nn - 1);
  const int nn = lane & 15;
  const int hi8 = (lane & 16) ? 8 : 0;

  const _Float16* xb  = xh + (size_t)b * Nn * Cc;
  const _Float16* ctr = xb + (size_t)n * Cc;
  const int nbr = idx[(size_t)pt * Kk + nn];
  const _Float16* nbp = xb + (size_t)nbr * Cc;
  float* outb = out + ((size_t)b * CHO) * Nn + n;

  // conv1 input: edge = [nbr - center (ch 0..63), center (ch 64..127)]
  v16h bf[4];
  bf[0] = load_b16_diff(nbp, ctr, 0, lane);
  bf[1] = load_b16_diff(nbp, ctr, 32, lane);
  bf[2] = load_b16(ctr, 0, lane);
  bf[3] = load_b16(ctr, 32, lane);

  // ---- conv1 + ReLU -> output channels [0,64), stage relu1 to LDS ----
#pragma unroll
  for (int m0 = 0; m0 < 4; ++m0) {
    v8f c;
#pragma unroll
    for (int r = 0; r < 8; ++r) c[r] = b1[m0 * 16 + hi8 + r];
#pragma unroll
    for (int kc = 0; kc < 4; ++kc)
      c = wmma_f16(load_a16(w1 + (size_t)(m0 * 16) * 128, 128, kc * 32, lane),
                   bf[kc], c);
#pragma unroll
    for (int r = 0; r < 8; ++r) c[r] = fmaxf(c[r], 0.0f);
#pragma unroll
    for (int r = 0; r < 8; ++r)
      y1s[wave][nn][m0 * 16 + hi8 + r] = (_Float16)c[r];
    warp_max_store(c, outb, m0 * 16, lane);
  }

  // ---- x passthrough channels [64,128): max over broadcast k == x itself ----
  {
    const float* xp = x + ((size_t)b * Cc) * Nn + n;
    for (int cch = lane; cch < Cc; cch += 32)
      outb[(size_t)(Cc + cch) * Nn] = xp[(size_t)cch * Nn];
  }
  __syncthreads();

  // ---- conv2 + ReLU -> channels [128,192): input = [relu1, center] ----
  v16h b2f0 = load_b16(&y1s[wave][nn][0], 0, lane);
  v16h b2f1 = load_b16(&y1s[wave][nn][0], 32, lane);
#pragma unroll
  for (int m0 = 0; m0 < 4; ++m0) {
    const _Float16* wr = w2 + (size_t)(m0 * 16) * 128;
    v8f c;
#pragma unroll
    for (int r = 0; r < 8; ++r) c[r] = b2[m0 * 16 + hi8 + r];
    c = wmma_f16(load_a16(wr, 128, 0, lane), b2f0, c);
    c = wmma_f16(load_a16(wr, 128, 32, lane), b2f1, c);
    c = wmma_f16(load_a16(wr, 128, 64, lane), bf[2], c);
    c = wmma_f16(load_a16(wr, 128, 96, lane), bf[3], c);
#pragma unroll
    for (int r = 0; r < 8; ++r) c[r] = fmaxf(c[r], 0.0f);
#pragma unroll
    for (int r = 0; r < 8; ++r)
      y2s[wave][nn][m0 * 16 + hi8 + r] = (_Float16)c[r];
    warp_max_store(c, outb, 128 + m0 * 16, lane);
  }
  __syncthreads();

  // ---- conv3 (no act) -> channels [192,256): input = [relu1, center, relu2] ----
  v16h b3f4 = load_b16(&y2s[wave][nn][0], 0, lane);
  v16h b3f5 = load_b16(&y2s[wave][nn][0], 32, lane);
#pragma unroll
  for (int m0 = 0; m0 < 4; ++m0) {
    const _Float16* wr = w3 + (size_t)(m0 * 16) * 192;
    v8f c;
#pragma unroll
    for (int r = 0; r < 8; ++r) c[r] = b3[m0 * 16 + hi8 + r];
    c = wmma_f16(load_a16(wr, 192, 0, lane), b2f0, c);
    c = wmma_f16(load_a16(wr, 192, 32, lane), b2f1, c);
    c = wmma_f16(load_a16(wr, 192, 64, lane), bf[2], c);
    c = wmma_f16(load_a16(wr, 192, 96, lane), bf[3], c);
    c = wmma_f16(load_a16(wr, 192, 128, lane), b3f4, c);
    c = wmma_f16(load_a16(wr, 192, 160, lane), b3f5, c);
    warp_max_store(c, outb, 192 + m0 * 16, lane);
  }
}

// ---------------- launch ----------------
extern "C" void kernel_launch(void* const* d_in, const int* in_sizes, int n_in,
                              void* d_out, int out_size, void* d_ws, size_t ws_size,
                              hipStream_t stream) {
  const float* x  = (const float*)d_in[0];
  const float* W1 = (const float*)d_in[1];
  const float* b1 = (const float*)d_in[2];
  const float* W2 = (const float*)d_in[3];
  const float* b2 = (const float*)d_in[4];
  const float* W3 = (const float*)d_in[5];
  const float* b3 = (const float*)d_in[6];

  char* ws = (char*)d_ws;
  _Float16* xh  = (_Float16*)(ws + OFF_XH);
  float*    xx  = (float*)(ws + OFF_XX);
  _Float16* w1h = (_Float16*)(ws + OFF_W1);
  _Float16* w2h = (_Float16*)(ws + OFF_W2);
  _Float16* w3h = (_Float16*)(ws + OFF_W3);
  int*      idx = (int*)(ws + OFF_IDX);
  float*    S   = (float*)(ws + OFF_S);
  float*    out = (float*)d_out;

  prep_x_kernel<<<Bn * Nn / 256, 256, 0, stream>>>(x, xh, xx);
  prep_w_kernel<<<112, 256, 0, stream>>>(W1, W2, W3, w1h, w2h, w3h);
  for (int b = 0; b < Bn; ++b) {
    // 128 row-tiles x 32 col-groups = 4096 strips, 8 waves/block
    dist_gemm_kernel<<<512, 256, 0, stream>>>(xh, S, b);
    // one wave per row: 2048 waves = 256 blocks
    topk_kernel<<<Nn / 8, 256, 0, stream>>>(S, xx, idx, b);
  }
  fused_dense_conv_kernel<<<Bn * Nn / 8, 256, 0, stream>>>(
      xh, idx, w1h, w2h, w3h, b1, b2, b3, x, out);
}
